// Attention_76175539962198
// MI455X (gfx1250) — matrix operand order repaired and checked by reading, not compiled, not run on hardware
//
#include <hip/hip_runtime.h>
#include <hip/hip_bf16.h>

// ---------------------------------------------------------------------------
// MHA forward for B=2,S=2048,D=2048,H=16,DH=128 on gfx1250 (CDNA5, wave32).
// All matmuls use v_wmma_f32_16x16x32_bf16 (bf16 in, f32 accumulate).
// Row-major tile staging uses CDNA5 async copies (global_load_async_to_lds_b128,
// ASYNCcnt); transposed staging goes through registers. Double-buffered LDS in
// both the GEMM and the flash-attention kernel (CDNA5 WGP LDS is 320 KB).
// Pipeline: cast/transpose -> 3x GEMM (Q,K,V) -> flash attention -> GEMM (out)
// Workspace usage: ~112 MiB (see offsets in kernel_launch).
// ---------------------------------------------------------------------------

typedef __bf16 bf16_t;
typedef __attribute__((ext_vector_type(16))) __bf16 v16bf;
typedef __attribute__((ext_vector_type(8)))  float  v8f;
typedef __attribute__((ext_vector_type(2)))  __bf16 bf16x2;
typedef __attribute__((ext_vector_type(4)))  __bf16 bf16x4;
typedef __attribute__((ext_vector_type(8)))  __bf16 bf16x8;

#define ATT_B  2
#define ATT_S  2048
#define ATT_D  2048
#define ATT_H  16
#define ATT_DH 128
#define MROWS  (ATT_B * ATT_S)          // 4096
#define KD     ATT_D                    // 2048 (GEMM K)
#define ND     (ATT_H * ATT_DH)         // 2048 (GEMM N)

// ---------------------------------------------------------------------------
// CDNA5 async memory->LDS copy (no VGPR round-trip, tracked by ASYNCcnt).
// VDST = per-lane LDS byte address (low 32 bits of the flat LDS pointer),
// VADDR = 64-bit global address, GV mode (saddr = off).
// ---------------------------------------------------------------------------
__device__ __forceinline__ void async_load_b128(const bf16_t* g, bf16_t* l) {
  unsigned lds = (unsigned)(unsigned long long)l;     // LDS byte address
  asm volatile("global_load_async_to_lds_b128 %0, %1, off"
               :: "v"(lds), "v"((unsigned long long)g) : "memory");
}
__device__ __forceinline__ void wait_async0() {
#if __has_builtin(__builtin_amdgcn_s_wait_asynccnt)
  __builtin_amdgcn_s_wait_asynccnt(0);
#else
  asm volatile("s_wait_asynccnt 0x0" ::: "memory");
#endif
}

// ---------------------------------------------------------------------------
// f32 -> bf16 cast (4 elems / thread)
// ---------------------------------------------------------------------------
__global__ __launch_bounds__(256) void k_cast_bf16(const float* __restrict__ src,
                                                   unsigned short* __restrict__ dstu) {
  bf16_t* dst = (bf16_t*)dstu;
  size_t i = ((size_t)blockIdx.x * blockDim.x + threadIdx.x) * 4;
  float4 v = *(const float4*)(src + i);
  bf16x4 o;
  o[0] = (bf16_t)v.x; o[1] = (bf16_t)v.y; o[2] = (bf16_t)v.z; o[3] = (bf16_t)v.w;
  *(bf16x4*)(dst + i) = o;
}

// ---------------------------------------------------------------------------
// W_{Q,K,V}: [H][D][DH] f32  ->  [D][H*DH] bf16 (so QKV becomes one flat GEMM)
// ---------------------------------------------------------------------------
__global__ __launch_bounds__(256) void k_transpose_w(const float* __restrict__ w,
                                                     unsigned short* __restrict__ dstu) {
  bf16_t* dst = (bf16_t*)dstu;
  int tid = blockIdx.x * blockDim.x + threadIdx.x;  // over D*H*DH = 4194304
  int d  = tid >> 11;        // / 2048
  int hk = tid & 2047;       // h*128 + k
  int h  = hk >> 7;
  int k  = hk & 127;
  float v = w[((size_t)h * ATT_D + d) * ATT_DH + k];
  dst[(size_t)d * ND + hk] = (bf16_t)v;
}

// ---------------------------------------------------------------------------
// GEMM: C[M][N] = A[M][K] * B[K][N] + column bias. A,B bf16; C bf16 or f32.
// Block = 256 threads (8 waves). Tile 128x128, k-step 32, LDS double-buffered.
// A-tile: async global->LDS copies. B-tile: register-staged transpose.
// Per wave: 16 M-rows x full 128 N => 8 x v8f accumulators.
// ---------------------------------------------------------------------------
#define GBM 128
#define GBN 128
#define GBK 32
#define GLDA 40
#define GLDB 36

template <bool OUTF32>
__global__ __launch_bounds__(256) void k_gemm_bf16(
    const unsigned short* __restrict__ Au, const unsigned short* __restrict__ Bu,
    unsigned short* __restrict__ Cbfu, float* __restrict__ Cf32,
    const float* __restrict__ bias, int M, int N, int K)
{
  const bf16_t* A  = (const bf16_t*)Au;
  const bf16_t* Bw = (const bf16_t*)Bu;
  bf16_t* Cbf = (bf16_t*)Cbfu;

  __shared__ bf16_t As[2][GBM * GLDA];   // 2 x 10240 B
  __shared__ bf16_t Bs[2][GBN * GLDB];   // 2 x  9216 B

  const int tid   = threadIdx.x;
  const int lane  = tid & 31;
  const int wave  = tid >> 5;
  const int l16   = lane & 15;
  const int khalf = lane >> 4;
  const int m0 = blockIdx.y * GBM;
  const int n0 = blockIdx.x * GBN;
  const int arow = wave * 16 + l16;

  // per-thread staging assignment: A chunks (j*64 + tid/4, (tid&3)*8),
  // B chunks (j*16 + tid/16, (tid&15)*8)
  const int ar0 = tid >> 2;           // A row base (0..63), +64 for j=1
  const int akc = (tid & 3) * 8;      // A k-chunk
  const int bk0 = tid >> 4;           // B k-row base (0..15), +16 for j=1
  const int bnc = (tid & 15) * 8;     // B n-chunk

  // hoisted column-bias (8 values per lane)
  float bv[8];
#pragma unroll
  for (int t = 0; t < 8; ++t) bv[t] = bias[n0 + t * 16 + l16];

  const v8f zerov = {0.f,0.f,0.f,0.f,0.f,0.f,0.f,0.f};
  v8f acc[8];
#pragma unroll
  for (int t = 0; t < 8; ++t) acc[t] = zerov;

  const int nk = K / GBK;
  bf16x8 breg[2];

  // ---- prologue: tile 0 -> buffer 0 (A async, B via regs) ----
#pragma unroll
  for (int j = 0; j < 2; ++j) {
    int r = j * 64 + ar0;
    async_load_b128(&A[(size_t)(m0 + r) * K + akc], &As[0][r * GLDA + akc]);
    breg[j] = *(const bf16x8*)&Bw[(size_t)(j * 16 + bk0) * N + n0 + bnc];
  }
#pragma unroll
  for (int j = 0; j < 2; ++j)
#pragma unroll
    for (int e = 0; e < 8; ++e)
      Bs[0][(bnc + e) * GLDB + (j * 16 + bk0)] = breg[j][e];
  wait_async0();
  __syncthreads();

  for (int i = 0; i < nk; ++i) {
    const int cur = i & 1;
    // ---- issue next tile's loads (in flight during the math below) ----
    if (i + 1 < nk) {
      const int k0n = (i + 1) * GBK;
#pragma unroll
      for (int j = 0; j < 2; ++j) {
        int r = j * 64 + ar0;
        async_load_b128(&A[(size_t)(m0 + r) * K + k0n + akc],
                        &As[cur ^ 1][r * GLDA + akc]);
        breg[j] = *(const bf16x8*)&Bw[(size_t)(k0n + j * 16 + bk0) * N + n0 + bnc];
      }
    }

    // ---- batch-load all fragments, then 8 back-to-back WMMAs ----
    const bf16_t* as = As[cur];
    const bf16_t* bs = Bs[cur];
    v16bf af;
    {
      const bf16_t* ab = &as[arow * GLDA];
#pragma unroll
      for (int q = 0; q < 8; ++q) {
        int kk = khalf * 16 + 2 * q;
        bf16x2 p = *(const bf16x2*)&ab[kk];
        af[2 * q] = p[0]; af[2 * q + 1] = p[1];
      }
    }
    v16bf bfr[8];
#pragma unroll
    for (int t = 0; t < 8; ++t) {
      const bf16_t* bb = &bs[(t * 16 + l16) * GLDB + khalf * 16];
#pragma unroll
      for (int q = 0; q < 8; ++q) {
        bf16x2 p = *(const bf16x2*)&bb[2 * q];
        bfr[t][2 * q] = p[0]; bfr[t][2 * q + 1] = p[1];
      }
    }
#pragma unroll
    for (int t = 0; t < 8; ++t)
      acc[t] = __builtin_amdgcn_wmma_f32_16x16x32_bf16(
          false, af, false, bfr[t], (short)0, acc[t], false, false);

    // ---- stage next B tile, drain async A, one barrier per k-step ----
    if (i + 1 < nk) {
      const int nxt = cur ^ 1;
#pragma unroll
      for (int j = 0; j < 2; ++j)
#pragma unroll
        for (int e = 0; e < 8; ++e)
          Bs[nxt][(bnc + e) * GLDB + (j * 16 + bk0)] = breg[j][e];
      wait_async0();
      __syncthreads();
    }
  }

  // ---- branch-free epilogue: C layout lane l16 = col, khalf*8 + r = row ----
#pragma unroll
  for (int t = 0; t < 8; ++t) {
    const int col = n0 + t * 16 + l16;
#pragma unroll
    for (int r = 0; r < 8; ++r) {
      const int row = m0 + wave * 16 + khalf * 8 + r;
      const float v = acc[t][r] + bv[t];
      if constexpr (OUTF32) Cf32[(size_t)row * N + col] = v;
      else                  Cbf[(size_t)row * N + col] = (bf16_t)v;
    }
  }
}

// ---------------------------------------------------------------------------
// Flash attention (causal). Q,K,V,Z: bf16 [B*S][H*DH] row-major.
// Grid (S/128, H, B); 256 threads = 8 waves; wave owns 16 q-rows.
// kv-tile = 128, K/V double-buffered (~171 KB LDS of the 320 KB WGP pool).
// K tile: async global->LDS (row-major). V tile: register-staged transpose.
// ---------------------------------------------------------------------------
#define FLK   128         // kv tile
#define LDSK  136         // K tile row stride (kv x d), rows 16B-aligned
#define LDSV  132         // V^T / P row stride

__global__ __launch_bounds__(256) void k_flash_attn(
    const unsigned short* __restrict__ Qu, const unsigned short* __restrict__ Ku,
    const unsigned short* __restrict__ Vu, unsigned short* __restrict__ Zu)
{
  const bf16_t* Q  = (const bf16_t*)Qu;
  const bf16_t* Kb = (const bf16_t*)Ku;
  const bf16_t* Vb = (const bf16_t*)Vu;
  bf16_t* Z = (bf16_t*)Zu;

  __shared__ bf16_t Ks[2][FLK * LDSK];     // 2 x 34816 B  [kv][d]
  __shared__ bf16_t Vt[2][ATT_DH * LDSV];  // 2 x 33792 B  [d][kv]
  __shared__ bf16_t Ps[128 * LDSV];        //     33792 B  [q][kv]

  const int tid   = threadIdx.x;
  const int lane  = tid & 31;
  const int wave  = tid >> 5;
  const int l16   = lane & 15;
  const int khalf = lane >> 4;

  const int qt = blockIdx.x;
  const int h  = blockIdx.y;
  const int bz = blockIdx.z;
  const int q0 = qt * 128;

  const size_t rowStride = ND;                 // 2048
  const size_t headOff   = (size_t)h * ATT_DH;
  const size_t baseRow   = (size_t)bz * ATT_S;

  const bf16_t* Kbase = Kb + baseRow * rowStride + headOff;
  const bf16_t* Vbase = Vb + baseRow * rowStride + headOff;

  // staging assignment: 8 chunks/thread, row j*16 + tid/16, d-chunk (tid&15)*8
  const int rb = tid >> 4;           // 0..15
  const int dc = (tid & 15) * 8;

  // ---- preload Q fragments (A layout) straight from global ----
  const int qrow = q0 + wave * 16 + l16;
  const bf16_t* qptr = Q + (baseRow + qrow) * rowStride + headOff;
  v16bf qf[4];
#pragma unroll
  for (int c = 0; c < 4; ++c) {
#pragma unroll
    for (int i = 0; i < 8; ++i) {
      int kk = khalf * 16 + 2 * i;
      bf16x2 p = *(const bf16x2*)&qptr[c * 32 + kk];
      qf[c][2 * i] = p[0]; qf[c][2 * i + 1] = p[1];
    }
  }

  const v8f zerov = {0.f,0.f,0.f,0.f,0.f,0.f,0.f,0.f};
  v8f o[8];
  float run_max[8], run_sum[8];
#pragma unroll
  for (int t = 0; t < 8; ++t) o[t] = zerov;
#pragma unroll
  for (int r = 0; r < 8; ++r) { run_max[r] = -1e30f; run_sum[r] = 0.f; }

  const float SCALE = 0.088388347648318447f;   // 1/sqrt(128)

  bf16x8 vreg[8];

  // ---- prologue: kv tile 0 -> buffer 0 (K async, V via regs) ----
#pragma unroll
  for (int j = 0; j < 8; ++j) {
    int r = j * 16 + rb;
    async_load_b128(Kbase + (size_t)r * rowStride + dc, &Ks[0][r * LDSK + dc]);
    vreg[j] = *(const bf16x8*)(Vbase + (size_t)r * rowStride + dc);
  }
#pragma unroll
  for (int j = 0; j < 8; ++j) {
    int r = j * 16 + rb;
#pragma unroll
    for (int e = 0; e < 8; ++e) Vt[0][(dc + e) * LDSV + r] = vreg[j][e];
  }
  wait_async0();
  __syncthreads();

  for (int kt = 0; kt <= qt; ++kt) {
    const int cur = kt & 1;
    // ---- issue next kv tile's loads (overlap with this tile's math) ----
    if (kt < qt) {
      const size_t nb = (size_t)(kt + 1) * FLK;
#pragma unroll
      for (int j = 0; j < 8; ++j) {
        int r = j * 16 + rb;
        async_load_b128(Kbase + (nb + r) * rowStride + dc,
                        &Ks[cur ^ 1][r * LDSK + dc]);
        vreg[j] = *(const bf16x8*)(Vbase + (nb + r) * rowStride + dc);
      }
    }
    const bf16_t* ks = Ks[cur];
    const bf16_t* vt = Vt[cur];

    // ---- scores S = Q * K^T : 8 kv-tiles of 16, paired fragment loads ----
    v8f s[8];
#pragma unroll
    for (int t = 0; t < 8; ++t) s[t] = zerov;
#pragma unroll
    for (int c = 0; c < 4; ++c) {          // d chunks of 32
#pragma unroll
      for (int t2 = 0; t2 < 8; t2 += 2) {
        v16bf b0, b1;                      // B = K^T: pairs contiguous in d
        const bf16_t* kb0 = &ks[((t2    ) * 16 + l16) * LDSK + c * 32 + khalf * 16];
        const bf16_t* kb1 = &ks[((t2 + 1) * 16 + l16) * LDSK + c * 32 + khalf * 16];
#pragma unroll
        for (int i = 0; i < 8; ++i) {
          bf16x2 p0 = *(const bf16x2*)&kb0[2 * i];
          bf16x2 p1 = *(const bf16x2*)&kb1[2 * i];
          b0[2 * i] = p0[0]; b0[2 * i + 1] = p0[1];
          b1[2 * i] = p1[0]; b1[2 * i + 1] = p1[1];
        }
        s[t2]     = __builtin_amdgcn_wmma_f32_16x16x32_bf16(
            false, qf[c], false, b0, (short)0, s[t2], false, false);
        s[t2 + 1] = __builtin_amdgcn_wmma_f32_16x16x32_bf16(
            false, qf[c], false, b1, (short)0, s[t2 + 1], false, false);
      }
    }

    // ---- scale (+ causal mask only on the diagonal tile) ----
    if (kt == qt) {
#pragma unroll
      for (int t = 0; t < 8; ++t) {
        int kvg = kt * FLK + t * 16 + l16;
#pragma unroll
        for (int r = 0; r < 8; ++r) {
          int qg = q0 + wave * 16 + khalf * 8 + r;
          float v = s[t][r] * SCALE;
          if (kvg > qg) v = -1e30f;
          s[t][r] = v;
        }
      }
    } else {
#pragma unroll
      for (int t = 0; t < 8; ++t)
#pragma unroll
        for (int r = 0; r < 8; ++r) s[t][r] *= SCALE;
    }

    // ---- online softmax (row reductions across 16-lane C groups) ----
#pragma unroll
    for (int r = 0; r < 8; ++r) {
      float m = -1e30f;
#pragma unroll
      for (int t = 0; t < 8; ++t) m = fmaxf(m, s[t][r]);
#pragma unroll
      for (int off = 1; off < 16; off <<= 1) m = fmaxf(m, __shfl_xor(m, off, 32));
      float nm = fmaxf(run_max[r], m);
      float corr = __expf(run_max[r] - nm);
      run_max[r] = nm;
      float ls = 0.f;
#pragma unroll
      for (int t = 0; t < 8; ++t) {
        float p = __expf(s[t][r] - nm);
        s[t][r] = p;
        ls += p;
      }
#pragma unroll
      for (int off = 1; off < 16; off <<= 1) ls += __shfl_xor(ls, off, 32);
      run_sum[r] = run_sum[r] * corr + ls;
#pragma unroll
      for (int t = 0; t < 8; ++t) o[t][r] *= corr;
    }

    // ---- P: C-layout regs -> LDS row-major (same-wave rows; DS in-order) ----
#pragma unroll
    for (int t = 0; t < 8; ++t)
#pragma unroll
      for (int r = 0; r < 8; ++r)
        Ps[(wave * 16 + khalf * 8 + r) * LDSV + t * 16 + l16] = (bf16_t)s[t][r];

    // ---- O += P * V ----
#pragma unroll
    for (int c = 0; c < 4; ++c) {          // kv chunks of 32
      v16bf pf;                            // A = P fragment
      const bf16_t* pb = &Ps[(wave * 16 + l16) * LDSV + c * 32];
#pragma unroll
      for (int i = 0; i < 8; ++i) {
        int kk = khalf * 16 + 2 * i;
        bf16x2 p = *(const bf16x2*)&pb[kk];
        pf[2 * i] = p[0]; pf[2 * i + 1] = p[1];
      }
#pragma unroll
      for (int t2 = 0; t2 < 8; t2 += 2) {  // d tiles of 16, paired loads
        v16bf b0, b1;                      // B = V: Vt is [d][kv] -> contiguous
        const bf16_t* vb0 = &vt[((t2    ) * 16 + l16) * LDSV + c * 32 + khalf * 16];
        const bf16_t* vb1 = &vt[((t2 + 1) * 16 + l16) * LDSV + c * 32 + khalf * 16];
#pragma unroll
        for (int i = 0; i < 8; ++i) {
          bf16x2 p0 = *(const bf16x2*)&vb0[2 * i];
          bf16x2 p1 = *(const bf16x2*)&vb1[2 * i];
          b0[2 * i] = p0[0]; b0[2 * i + 1] = p0[1];
          b1[2 * i] = p1[0]; b1[2 * i + 1] = p1[1];
        }
        o[t2]     = __builtin_amdgcn_wmma_f32_16x16x32_bf16(
            false, pf, false, b0, (short)0, o[t2], false, false);
        o[t2 + 1] = __builtin_amdgcn_wmma_f32_16x16x32_bf16(
            false, pf, false, b1, (short)0, o[t2 + 1], false, false);
      }
    }

    // ---- stage next V tile, drain async K, one barrier per kv-step ----
    if (kt < qt) {
      const int nxt = cur ^ 1;
#pragma unroll
      for (int j = 0; j < 8; ++j) {
        int r = j * 16 + rb;
#pragma unroll
        for (int e = 0; e < 8; ++e) Vt[nxt][(dc + e) * LDSV + r] = vreg[j][e];
      }
      wait_async0();
      __syncthreads();
    }
  }

  // ---- normalize and store z (bf16, [B,S,H,DH] == [B*S][2048]) ----
#pragma unroll
  for (int t = 0; t < 8; ++t) {
#pragma unroll
    for (int r = 0; r < 8; ++r) {
      int row = q0 + wave * 16 + khalf * 8 + r;
      float v = o[t][r] / run_sum[r];
      Z[(baseRow + row) * rowStride + headOff + t * 16 + l16] = (bf16_t)v;
    }
  }
}

// ---------------------------------------------------------------------------
// Host launcher
// ---------------------------------------------------------------------------
extern "C" void kernel_launch(void* const* d_in, const int* in_sizes, int n_in,
                              void* d_out, int out_size, void* d_ws, size_t ws_size,
                              hipStream_t stream) {
  const float* resid = (const float*)d_in[0];
  const float* W_Q   = (const float*)d_in[1];
  const float* W_K   = (const float*)d_in[2];
  const float* W_V   = (const float*)d_in[3];
  const float* W_O   = (const float*)d_in[4];
  const float* b_Q   = (const float*)d_in[5];
  const float* b_K   = (const float*)d_in[6];
  const float* b_V   = (const float*)d_in[7];
  const float* b_O   = (const float*)d_in[8];
  float* out = (float*)d_out;

  char* ws = (char*)d_ws;
  size_t off = 0;
  const size_t ACT = (size_t)MROWS * ND * 2;   // 16 MiB bf16 activation
  const size_t WGT = (size_t)KD * ND * 2;      //  8 MiB bf16 weight
  unsigned short* Xbf = (unsigned short*)(ws + off); off += ACT;
  unsigned short* WqB = (unsigned short*)(ws + off); off += WGT;
  unsigned short* WkB = (unsigned short*)(ws + off); off += WGT;
  unsigned short* WvB = (unsigned short*)(ws + off); off += WGT;
  unsigned short* WoB = (unsigned short*)(ws + off); off += WGT;
  unsigned short* Qb  = (unsigned short*)(ws + off); off += ACT;
  unsigned short* Kbf = (unsigned short*)(ws + off); off += ACT;
  unsigned short* Vbf = (unsigned short*)(ws + off); off += ACT;
  unsigned short* Zb  = (unsigned short*)(ws + off); off += ACT;  // ~112 MiB total

  // 1) casts / weight transposes
  k_cast_bf16   <<<(MROWS * (size_t)ATT_D) / 1024, 256, 0, stream>>>(resid, Xbf);
  k_transpose_w <<<(ATT_H * ATT_D * ATT_DH) / 256, 256, 0, stream>>>(W_Q, WqB);
  k_transpose_w <<<(ATT_H * ATT_D * ATT_DH) / 256, 256, 0, stream>>>(W_K, WkB);
  k_transpose_w <<<(ATT_H * ATT_D * ATT_DH) / 256, 256, 0, stream>>>(W_V, WvB);
  k_cast_bf16   <<<((size_t)KD * ND) / 1024, 256, 0, stream>>>(W_O, WoB);

  // 2) QKV projections: [4096x2048] x [2048x2048] -> bf16 [B,S,H,DH]
  dim3 ggrid(ND / GBN, MROWS / GBM);
  k_gemm_bf16<false><<<ggrid, 256, 0, stream>>>(Xbf, WqB, Qb,  nullptr, b_Q, MROWS, ND, KD);
  k_gemm_bf16<false><<<ggrid, 256, 0, stream>>>(Xbf, WkB, Kbf, nullptr, b_K, MROWS, ND, KD);
  k_gemm_bf16<false><<<ggrid, 256, 0, stream>>>(Xbf, WvB, Vbf, nullptr, b_V, MROWS, ND, KD);

  // 3) causal flash attention -> Z bf16 [B,S,H,DH]
  k_flash_attn<<<dim3(ATT_S / 128, ATT_H, ATT_B), 256, 0, stream>>>(Qb, Kbf, Vbf, Zb);

  // 4) output projection: Z x W_O + b_O -> f32 out
  k_gemm_bf16<true><<<ggrid, 256, 0, stream>>>(Zb, WoB, nullptr, out, b_O, MROWS, ATT_D, ND);
}
